// GNNStack_36137854828757
// MI455X (gfx1250) — compile-verified, matrix operand order checked by reading, not run on hardware
//
#include <hip/hip_runtime.h>
#include <hip/hip_bf16.h>

#define N_NODES 50000
#define N_EDGES 800000
#define F_IN 64
#define HID 128
#define N_GRAPHS 500
#define N_CLS 10
#define MLP_HID 50
#define KTOT 128                 // concatenated K = F_IN(agg) + F_IN(x)
#define M_TILES (N_NODES / 16)   // 3125, exact

typedef __attribute__((ext_vector_type(16))) __bf16 v16bf;
typedef __attribute__((ext_vector_type(8)))  float  v8f;

union Frag { unsigned u[8]; v16bf v; };

// packed f32x2 -> bf16x2 (RNE); lowers to v_cvt_pk_bf16_f32 on gfx1250
__device__ __forceinline__ unsigned pack2bf(float lo, float hi) {
  __hip_bfloat162 p = __float22bfloat162_rn(make_float2(lo, hi));
  unsigned u;
  __builtin_memcpy(&u, &p, sizeof(u));
  return u;
}

__device__ __forceinline__ unsigned short f2bf16_bits(float f) {
  __hip_bfloat16 b = __float2bfloat16(f);
  unsigned short u;
  __builtin_memcpy(&u, &b, sizeof(u));
  return u;
}

// ---------------------------------------------------------------------------
// Kernel: bf16 transposed concatenated weight  Wt[n][k], k in [0,128)
// ---------------------------------------------------------------------------
__global__ void prep_w_kernel(const float* __restrict__ Wn,
                              const float* __restrict__ Ws,
                              unsigned short* __restrict__ Wt) {
  int tid = blockIdx.x * blockDim.x + threadIdx.x;
  if (tid >= HID * KTOT) return;
  int n = tid >> 7;
  int k = tid & 127;
  float w = (k < F_IN) ? Wn[k * HID + n] : Ws[(k - F_IN) * HID + n];
  Wt[n * KTOT + k] = f2bf16_bits(w);
}

// ---------------------------------------------------------------------------
// CSR build pass 1: per-destination degree counts (int atomics)
// ---------------------------------------------------------------------------
__global__ void deg_count_kernel(const int* __restrict__ ei,
                                 int* __restrict__ degcnt) {
  int e = blockIdx.x * blockDim.x + threadIdx.x;
  if (e >= N_EDGES) return;
  atomicAdd(&degcnt[ei[N_EDGES + e]], 1);
}

// ---------------------------------------------------------------------------
// CSR build pass 2: exclusive scan of 50000 counts -> rowptr[50001]
// ---------------------------------------------------------------------------
__global__ void scan_kernel(const int* __restrict__ degcnt,
                            int* __restrict__ rowptr) {
  __shared__ int part[256];
  __shared__ int offs[256];
  int t = threadIdx.x;
  const int CH = (N_NODES + 255) / 256;          // 196
  int lo = t * CH;
  int hi = lo + CH < N_NODES ? lo + CH : N_NODES;
  int s = 0;
  for (int i = lo; i < hi; ++i) s += degcnt[i];
  part[t] = s;
  __syncthreads();
  if (t == 0) {
    int run = 0;
    for (int i = 0; i < 256; ++i) { offs[i] = run; run += part[i]; }
  }
  __syncthreads();
  int run = offs[t];
  for (int i = lo; i < hi; ++i) { rowptr[i] = run; run += degcnt[i]; }
  if (t == 255) rowptr[N_NODES] = run;           // == N_EDGES
}

// ---------------------------------------------------------------------------
// CSR build pass 3: scatter edge sources into CSR adjacency
// ---------------------------------------------------------------------------
__global__ void csr_fill_kernel(const int* __restrict__ ei,
                                const int* __restrict__ rowptr,
                                int* __restrict__ fill,
                                int* __restrict__ csr) {
  int e = blockIdx.x * blockDim.x + threadIdx.x;
  if (e >= N_EDGES) return;
  int dst = ei[N_EDGES + e];
  int pos = rowptr[dst] + atomicAdd(&fill[dst], 1);
  csr[pos] = ei[e];
}

// ---------------------------------------------------------------------------
// Gather-aggregate: one wave per node, lane f handles features 2f, 2f+1.
// Neighbor loop unrolled x4 to keep ~1KB of clean float2 loads in flight.
// ---------------------------------------------------------------------------
__global__ void gather_agg_kernel(const float* __restrict__ x,
                                  const int* __restrict__ rowptr,
                                  const int* __restrict__ csr,
                                  float* __restrict__ agg) {
  int gtid = blockIdx.x * blockDim.x + threadIdx.x;
  int node = gtid >> 5;
  if (node >= N_NODES) return;                   // wave-uniform exit
  int lane = threadIdx.x & 31;
  int beg = rowptr[node];
  int end = rowptr[node + 1];
  float ax = 0.f, ay = 0.f;
  int e = beg;
  for (; e + 4 <= end; e += 4) {
    int s0 = csr[e + 0], s1 = csr[e + 1], s2 = csr[e + 2], s3 = csr[e + 3];
    float2 v0 = *(const float2*)(x + (size_t)s0 * F_IN + 2 * lane);
    float2 v1 = *(const float2*)(x + (size_t)s1 * F_IN + 2 * lane);
    float2 v2 = *(const float2*)(x + (size_t)s2 * F_IN + 2 * lane);
    float2 v3 = *(const float2*)(x + (size_t)s3 * F_IN + 2 * lane);
    ax += v0.x + v1.x + v2.x + v3.x;
    ay += v0.y + v1.y + v2.y + v3.y;
  }
  for (; e < end; ++e) {
    int s = csr[e];
    float2 v = *(const float2*)(x + (size_t)s * F_IN + 2 * lane);
    ax += v.x;
    ay += v.y;
  }
  float inv = 1.0f / fmaxf((float)(end - beg), 1.0f);
  float2 r = make_float2(ax * inv, ay * inv);
  *(float2*)(agg + (size_t)node * F_IN + 2 * lane) = r;
}

// ---------------------------------------------------------------------------
// Per-graph node counts
// ---------------------------------------------------------------------------
__global__ void graph_cnt_kernel(const int* __restrict__ batch,
                                 float* __restrict__ cnt) {
  int tid = blockIdx.x * blockDim.x + threadIdx.x;
  if (tid >= N_NODES) return;
  atomicAdd(&cnt[batch[tid]], 1.0f);
}

// ---------------------------------------------------------------------------
// Fused SAGE GEMM with WMMA:  emb = [agg | x] @ Wt^T + b  (M=50000,N=128,K=128)
// one wave per 16-row tile; 8 N-tiles x 4 K-steps of v_wmma_f32_16x16x32_bf16
// with double-buffered B fragments. Epilogue: store emb, relu, pooled atomics.
// ---------------------------------------------------------------------------
__global__ void sage_wmma_kernel(const float* __restrict__ x,
                                 const float* __restrict__ agg,
                                 const unsigned short* __restrict__ Wt,
                                 const float* __restrict__ bias,
                                 const int* __restrict__ batch,
                                 float* __restrict__ emb,
                                 float* __restrict__ pool) {
  int gtid = blockIdx.x * blockDim.x + threadIdx.x;
  int m_tile = gtid >> 5;
  if (m_tile >= M_TILES) return;                 // wave-uniform exit

  int lane  = threadIdx.x & 31;
  int laneM = lane & 15;                         // M (A) / N (B) within tile
  int hiK   = (lane >> 4) << 3;                  // +8 K-offset, lanes 16..31

  int rowA = m_tile * 16 + laneM;

  v8f acc[8];
#pragma unroll
  for (int n = 0; n < 8; ++n) acc[n] = (v8f){0.f,0.f,0.f,0.f,0.f,0.f,0.f,0.f};

  const unsigned* WtU = (const unsigned*)Wt;     // bf16 K-pairs, 4B aligned

#pragma unroll
  for (int ks = 0; ks < 4; ++ks) {
    // ---- A fragment (16x32 bf16, ISA VGPR layout); agg pre-normalized -----
    const float* abase = (ks < 2) ? (agg + (size_t)rowA * F_IN + ks * 32)
                                  : (x   + (size_t)rowA * F_IN + (ks - 2) * 32);
    Frag af;
#pragma unroll
    for (int v = 0; v < 8; ++v) {
      int kb = ((v >> 2) << 4) + hiK + ((v & 3) << 1);
      af.u[v] = pack2bf(abase[kb], abase[kb + 1]);
    }
    // ---- 8 N-tiles, double-buffered B fragments ---------------------------
    Frag bcur;
#pragma unroll
    for (int v = 0; v < 8; ++v) {
      int kb = ((v >> 2) << 4) + hiK + ((v & 3) << 1);
      bcur.u[v] = WtU[(laneM * KTOT + ks * 32 + kb) >> 1];
    }
#pragma unroll
    for (int n = 0; n < 8; ++n) {
      Frag bnxt;
      if (n < 7) {
        int ncol = (n + 1) * 16 + laneM;
#pragma unroll
        for (int v = 0; v < 8; ++v) {
          int kb = ((v >> 2) << 4) + hiK + ((v & 3) << 1);
          bnxt.u[v] = WtU[(ncol * KTOT + ks * 32 + kb) >> 1];
        }
      }
      acc[n] = __builtin_amdgcn_wmma_f32_16x16x32_bf16(
          false, af.v, false, bcur.v, (short)0, acc[n], false, false);
      if (n < 7) bcur = bnxt;
    }
  }

  // ---- epilogue: bias, store emb, relu + pooled atomic accumulation -------
  int brow[8];
#pragma unroll
  for (int r = 0; r < 8; ++r)
    brow[r] = batch[m_tile * 16 + r + ((lane >> 4) << 3)];

#pragma unroll
  for (int n = 0; n < 8; ++n) {
    int col = n * 16 + laneM;
    float bv = bias[col];
#pragma unroll
    for (int r = 0; r < 8; ++r) {
      int rowg = m_tile * 16 + r + ((lane >> 4) << 3);
      float val = acc[n][r] + bv;
      emb[(size_t)rowg * HID + col] = val;
      float h = fmaxf(val, 0.0f);
      atomicAdd(&pool[(size_t)brow[r] * HID + col], h);
    }
  }
}

// ---------------------------------------------------------------------------
// Per-graph mean pool + MLP(128->50->10) + log_softmax
// ---------------------------------------------------------------------------
__global__ void mlp_head_kernel(const float* __restrict__ pool,
                                const float* __restrict__ cnt,
                                const float* __restrict__ W1,
                                const float* __restrict__ b1,
                                const float* __restrict__ W2,
                                const float* __restrict__ b2,
                                float* __restrict__ out) {
  __shared__ float sp[HID];
  __shared__ float sh[MLP_HID];
  __shared__ float sl[N_CLS];
  int g = blockIdx.x;
  int t = threadIdx.x;
  float invc = 1.0f / fmaxf(cnt[g], 1.0f);
  if (t < HID) sp[t] = pool[(size_t)g * HID + t] * invc;
  __syncthreads();
  if (t < MLP_HID) {
    float a = b1[t];
#pragma unroll 8
    for (int k = 0; k < HID; ++k) a += sp[k] * W1[k * MLP_HID + t];
    sh[t] = fmaxf(a, 0.0f);
  }
  __syncthreads();
  if (t < N_CLS) {
    float a = b2[t];
#pragma unroll
    for (int j = 0; j < MLP_HID; ++j) a += sh[j] * W2[j * N_CLS + t];
    sl[t] = a;
  }
  __syncthreads();
  if (t == 0) {
    float m = -1e30f;
    for (int i = 0; i < N_CLS; ++i) m = fmaxf(m, sl[i]);
    float s = 0.0f;
    for (int i = 0; i < N_CLS; ++i) s += __expf(sl[i] - m);
    float lse = __logf(s);
    for (int i = 0; i < N_CLS; ++i) out[g * N_CLS + i] = sl[i] - m - lse;
  }
}

// ---------------------------------------------------------------------------
extern "C" void kernel_launch(void* const* d_in, const int* in_sizes, int n_in,
                              void* d_out, int out_size, void* d_ws, size_t ws_size,
                              hipStream_t stream) {
  const float* x     = (const float*)d_in[0];
  const int*   ei    = (const int*)d_in[1];
  const int*   batch = (const int*)d_in[2];
  const float* Wn    = (const float*)d_in[3];
  const float* Ws    = (const float*)d_in[4];
  const float* bvec  = (const float*)d_in[5];
  const float* W1    = (const float*)d_in[6];
  const float* b1    = (const float*)d_in[7];
  const float* W2    = (const float*)d_in[8];
  const float* b2    = (const float*)d_in[9];
  float* out = (float*)d_out;

  // ---- workspace layout: zeroed region first ------------------------------
  int*   degcnt = (int*)d_ws;                              // 50000   (zeroed)
  int*   fill   = degcnt + N_NODES;                        // 50000   (zeroed)
  float* pool   = (float*)(fill + N_NODES);                // 500*128 (zeroed)
  float* cnt    = pool + (size_t)N_GRAPHS * HID;           // 500     (zeroed)
  int*   rowptr = (int*)(cnt + N_GRAPHS);                  // 50001
  int*   csr    = rowptr + (N_NODES + 1);                  // 800000
  float* agg    = (float*)(csr + N_EDGES);                 // 50000*64
  unsigned short* Wt = (unsigned short*)(agg + (size_t)N_NODES * F_IN); // 128*128

  size_t zeroBytes = (size_t)(N_NODES + N_NODES) * sizeof(int) +
                     (size_t)(N_GRAPHS * HID + N_GRAPHS) * sizeof(float);
  (void)hipMemsetAsync(d_ws, 0, zeroBytes, stream);

  prep_w_kernel    <<<(HID * KTOT + 255) / 256, 256, 0, stream>>>(Wn, Ws, Wt);
  deg_count_kernel <<<(N_EDGES + 255) / 256, 256, 0, stream>>>(ei, degcnt);
  scan_kernel      <<<1, 256, 0, stream>>>(degcnt, rowptr);
  csr_fill_kernel  <<<(N_EDGES + 255) / 256, 256, 0, stream>>>(ei, rowptr, fill, csr);
  gather_agg_kernel<<<(N_NODES * 32 + 127) / 128, 128, 0, stream>>>(x, rowptr, csr, agg);
  graph_cnt_kernel <<<(N_NODES + 255) / 256, 256, 0, stream>>>(batch, cnt);
  sage_wmma_kernel <<<(M_TILES + 3) / 4, 128, 0, stream>>>(
      x, agg, Wt, bvec, batch, out, pool);
  mlp_head_kernel  <<<N_GRAPHS, 128, 0, stream>>>(
      pool, cnt, W1, b1, W2, b2, out + (size_t)N_NODES * HID);
}
